// LoraConv2dExperts_72206990180988
// MI455X (gfx1250) — matrix operand order repaired and checked by reading, not compiled, not run on hardware
//
#include <hip/hip_runtime.h>

// ---------------------------------------------------------------------------
// LoRA-MoE conv  ==  two stacked GEMMs on CDNA5 WMMA (wave32, 16x16x32 f16)
//   Stage1: Ht[N, 64]  = (Wi[64, 768] x Xcol[768, N])^T   (1x3 conv, im2col W)
//   Stage2: out[256,N] = Wo[256,192]  x  Hcol[192, N]     (3x1 conv, im2col H)
// probs folded into Wo.  N = B*H*W = 65536.  Ht kept in f16 (8 MB, L2).
// Ht is n-major ([n][m], 64 rank-expert rows contiguous) so:
//   - stage-1 C fragments store as one b128 per lane (8 contiguous m),
//   - stage-2 B-fragment staging is a pure 16B-chunk copy -> async to LDS.
// Stage-1 uses a width-halo in LDS: each x block read from HBM exactly once,
// 3 conv taps = shifted LDS reads, no branches in the hot loop.
// ---------------------------------------------------------------------------

typedef __attribute__((ext_vector_type(16))) _Float16 v16h;
typedef __attribute__((ext_vector_type(8)))  _Float16 v8h;
typedef __attribute__((ext_vector_type(8)))  float    v8f;

#define E_    8
#define R_    8
#define CIN_  256
#define COUT_ 256
#define NB_   16
#define NH_   64
#define NW_   64
#define NTOT  (NB_ * NH_ * NW_)   // 65536
#define K1_   768                 // CIN * 3
#define K2_   192                 // 3 * E * R
#define M1_   64                  // E * R

// Build a 16-bit A/B fragment from two 8-half (16-byte) LDS chunks.
__device__ __forceinline__ v16h frag2(const _Float16* p0, const _Float16* p1) {
  const v8h a = *(const v8h*)p0;
  const v8h b = *(const v8h*)p1;
  v16h r;
#pragma unroll
  for (int i = 0; i < 8; ++i) { r[i] = a[i]; r[i + 8] = b[i]; }
  return r;
}

// CDNA5 async 16B global -> LDS copy (tracked by ASYNCcnt, no VGPR roundtrip).
// Generic pointers to __shared__ carry the LDS byte offset in their low 32 bits.
__device__ __forceinline__ void async_cp16(_Float16* lds_dst,
                                           const _Float16* gsrc) {
  const unsigned lds_off = (unsigned)(unsigned long long)lds_dst;
  asm volatile("global_load_async_to_lds_b128 %0, %1, off"
               :: "v"(lds_off), "v"(gsrc) : "memory");
}
__device__ __forceinline__ void wait_async0() {
  asm volatile("s_wait_asynccnt 0x0" ::: "memory");
}

__device__ __forceinline__ v8f wmma_f16(const v16h a, const v16h b, v8f c) {
  return __builtin_amdgcn_wmma_f32_16x16x32_f16(false, a, false, b,
                                                (short)0, c, false, false);
}

// ---------------------------------------------------------------------------
// Weight prep: f32 -> f16, GEMM layout, probs folded into Wo.
//   Wi[m=e*8+r][k=dx*256+c]   = weight_in[e,r,c,0,dx]
//   Wo[o][k2=dy*64+e*8+r]     = probs[e] * weight_out[e,o,r,dy,0]
// ---------------------------------------------------------------------------
__global__ void lora_prep_kernel(const float* __restrict__ probs,
                                 const float* __restrict__ win,
                                 const float* __restrict__ wout,
                                 _Float16* __restrict__ Wi,
                                 _Float16* __restrict__ Wo) {
  const int idx = blockIdx.x * blockDim.x + threadIdx.x;
  if (idx < M1_ * K1_) {
    const int m = idx / K1_, k = idx % K1_;
    const int dx = k / CIN_, c = k % CIN_;
    Wi[idx] = (_Float16)win[(m * CIN_ + c) * 3 + dx];
  } else if (idx < M1_ * K1_ + COUT_ * K2_) {
    const int j  = idx - M1_ * K1_;
    const int o  = j / K2_, k2 = j % K2_;
    const int dy = k2 / M1_, m = k2 % M1_;
    const int e  = m >> 3,   r = m & 7;
    Wo[j] = (_Float16)(probs[e] * wout[((e * COUT_ + o) * R_ + r) * 3 + dy]);
  }
}

// ---------------------------------------------------------------------------
// Stage 1: Ht[n][m] = (Wi x Xcol)^T.  Block = 256 thr (8 waves) -> 64 m x
// 64 n tile (one spatial row: fixed b,y).  Wave (m_strip, n_half) owns 16x32.
// Loop over 8 channel blocks of 32; each block: stage x row (+halo) once,
// then 3 shifted taps (dx) -> 6 WMMAs per wave.
// ---------------------------------------------------------------------------
__global__ __launch_bounds__(256)
void lora_stage1_kernel(const float* __restrict__ x,
                        const _Float16* __restrict__ Wi,
                        _Float16* __restrict__ Ht) {
  __shared__ __align__(32) _Float16 wi_tile[M1_ * 96];  // 12 KB [m][dx*32+cc]
  __shared__ __align__(32) _Float16 x_halo[66 * 32];    // 4.1 KB [xcol+1][cc]

  const int t    = threadIdx.x;
  const int lane = t & 31;
  const int wid  = t >> 5;
  const int blk  = blockIdx.x;          // blk = b*64 + y
  const int b    = blk >> 6;
  const int y    = blk & 63;
  const int n_base = blk * NW_;

  const int m_strip = wid & 3;          // 4 strips of 16 rows
  const int n_half  = wid >> 2;         // 2 halves of 32 cols
  const int m_base  = m_strip * 16;
  const int lh  = lane >> 4;            // 0/1 half-wave
  const int l15 = lane & 15;

  // staging assignments
  const int xcol = t & 63;              // spatial column within row
  const int cg   = t >> 6;              // channel group 0..3 (8 ch each)
  const int wm   = t >> 2;              // Wi row 0..63
  const int wsub = t & 3;               // 8-half sub-chunk within 32 channels

  // zero the two halo columns once (stay zero across channel blocks)
  if (t < 8) {
    v8h z = {};
    const int row = (t & 4) ? 65 : 0;
    *(v8h*)&x_halo[row * 32 + (t & 3) * 8] = z;
  }

  v8f acc0 = {}, acc1 = {};

  for (int cb = 0; cb < 8; ++cb) {
    const int c0 = cb * 32;
    __syncthreads();
    // Wi chunk for this channel block, all 3 taps: [64 m][3 dx][32 cc].
    // Async global->LDS, overlapped with the x staging below.
#pragma unroll
    for (int dx = 0; dx < 3; ++dx)
      async_cp16(&wi_tile[wm * 96 + dx * 32 + wsub * 8],
                 &Wi[wm * K1_ + dx * CIN_ + c0 + wsub * 8]);
    // x interior: 8 strided f32 channel loads per thread, cvt to f16, one
    // 16B LDS store.  Unconditional (halo handles the conv padding).
    {
      const float* xp = x + (size_t)(b * CIN_ + c0 + cg * 8) * (NH_ * NW_)
                          + y * NW_ + xcol;
      v8h v;
#pragma unroll
      for (int i = 0; i < 8; ++i)
        v[i] = (_Float16)xp[(size_t)i * (NH_ * NW_)];
      *(v8h*)&x_halo[(xcol + 1) * 32 + cg * 8] = v;
    }
    wait_async0();
    __syncthreads();

#pragma unroll
    for (int dx = 0; dx < 3; ++dx) {
      // A: lane<16 -> K {0..7,16..23}, lane>=16 -> K {8..15,24..31}
      const int arow = m_base + l15;
      const v16h afrag = frag2(&wi_tile[arow * 96 + dx * 32 + lh * 8],
                               &wi_tile[arow * 96 + dx * 32 + lh * 8 + 16]);
      // B: same K split as A; halo-shifted column = n + dx
      const int bbase = lh * 8;
      const int nc0 = n_half * 32 + l15 + dx;
      acc0 = wmma_f16(afrag,
                      frag2(&x_halo[nc0 * 32 + bbase],
                            &x_halo[nc0 * 32 + bbase + 16]), acc0);
      const int nc1 = nc0 + 16;
      acc1 = wmma_f16(afrag,
                      frag2(&x_halo[nc1 * 32 + bbase],
                            &x_halo[nc1 * 32 + bbase + 16]), acc1);
    }
  }

  // Store Ht[n][m] in f16.  C layout: element i is m = m_base + 8*lh + i,
  // n = l15 -> each lane owns 8 contiguous m values: one b128 store per tile.
#pragma unroll
  for (int j = 0; j < 2; ++j) {
    const v8f acc = j ? acc1 : acc0;
    const int n_g = n_base + n_half * 32 + j * 16 + l15;
    v8h hv;
#pragma unroll
    for (int i = 0; i < 8; ++i) hv[i] = (_Float16)acc[i];
    *(v8h*)&Ht[(size_t)n_g * M1_ + m_base + lh * 8] = hv;
  }
}

// ---------------------------------------------------------------------------
// Stage 2: out[256, N] = Wo x Hcol.  Block = 256 thr -> 256x64 tile (one
// spatial row).  Wave owns 32 output channels x 64 cols = 8 WMMA tiles.
// k-loop: 6 steps of 32 (each step = one dy, 32 of the 64 (e,r) rows).
// Both tiles staged with async global->LDS (Ht layout needs no transpose).
// ---------------------------------------------------------------------------
__global__ __launch_bounds__(256)
void lora_stage2_kernel(const _Float16* __restrict__ Ht,
                        const _Float16* __restrict__ Wo,
                        float* __restrict__ out) {
  __shared__ __align__(32) _Float16 wo_tile[COUT_ * 32];  // 16 KB [o][kk]
  __shared__ __align__(32) _Float16 h_tile[NW_ * 32];     //  4 KB [n][kk]

  const int t    = threadIdx.x;
  const int lane = t & 31;
  const int wid  = t >> 5;
  const int blk  = blockIdx.x;
  const int b    = blk >> 6;
  const int y    = blk & 63;

  const int lh  = lane >> 4;
  const int l15 = lane & 15;

  const int hn  = t & 63;   // col for H staging
  const int hkg = t >> 6;   // k-group 0..3

  v8f acc[2][4];
#pragma unroll
  for (int s = 0; s < 2; ++s)
#pragma unroll
    for (int j = 0; j < 4; ++j) { v8f z = {}; acc[s][j] = z; }

  for (int ks = 0; ks < 6; ++ks) {
    const int  dy = ks >> 1;
    const int  m0 = (ks & 1) * 32;
    const int  ry = y + dy - 1;               // shifted row (3x1 conv, pad 1)
    const bool rv = (ry >= 0) && (ry < NH_);  // uniform over the block

    __syncthreads();
    // Wo tile: thread t async-copies the 32 halves of output-channel row o=t
#pragma unroll
    for (int j = 0; j < 4; ++j)
      async_cp16(&wo_tile[t * 32 + j * 8],
                 &Wo[t * K2_ + ks * 32 + j * 8]);
    // H tile [n][kk]: straight 16B-chunk copy from Ht (async); rows outside
    // the image (uniform condition) are zero-filled in LDS instead.
    if (rv) {
      const size_t n_src = (size_t)(b * NH_ + ry) * NW_ + hn;
      async_cp16(&h_tile[hn * 32 + hkg * 8],
                 &Ht[n_src * M1_ + m0 + hkg * 8]);
    } else {
      v8h z = {};
      *(v8h*)&h_tile[hn * 32 + hkg * 8] = z;
    }
    wait_async0();
    __syncthreads();

#pragma unroll
    for (int s = 0; s < 2; ++s) {
      const int mrow = wid * 32 + s * 16 + l15;
      const v16h afrag = frag2(&wo_tile[mrow * 32 + lh * 8],
                               &wo_tile[mrow * 32 + lh * 8 + 16]);
#pragma unroll
      for (int j = 0; j < 4; ++j) {
        const int ncol = j * 16 + l15;
        const v16h bfrag = frag2(&h_tile[ncol * 32 + lh * 8],
                                 &h_tile[ncol * 32 + lh * 8 + 16]);
        acc[s][j] = wmma_f16(afrag, bfrag, acc[s][j]);
      }
    }
  }

  // f32 stores straight from C fragments (coalesced along x).
#pragma unroll
  for (int s = 0; s < 2; ++s)
#pragma unroll
    for (int j = 0; j < 4; ++j)
#pragma unroll
      for (int i = 0; i < 8; ++i) {
        const int o  = wid * 32 + s * 16 + i + lh * 8;
        const int xc = j * 16 + l15;
        out[(size_t)((b * COUT_ + o) * NH_ + y) * NW_ + xc] = acc[s][j][i];
      }
}

// ---------------------------------------------------------------------------
extern "C" void kernel_launch(void* const* d_in, const int* in_sizes, int n_in,
                              void* d_out, int out_size, void* d_ws, size_t ws_size,
                              hipStream_t stream) {
  (void)in_sizes; (void)n_in; (void)out_size; (void)ws_size;
  const float* x     = (const float*)d_in[0];
  const float* probs = (const float*)d_in[1];
  const float* win   = (const float*)d_in[2];
  const float* wout  = (const float*)d_in[3];
  float* out = (float*)d_out;

  // Workspace: [Wi f16 96KB][Wo f16 96KB] ... @256KB: [Ht f16 8MB]
  _Float16* Wi = (_Float16*)d_ws;
  _Float16* Wo = Wi + M1_ * K1_;
  _Float16* Ht = (_Float16*)((char*)d_ws + (256u << 10));

  lora_prep_kernel<<<dim3((M1_ * K1_ + COUT_ * K2_ + 255) / 256), dim3(256),
                     0, stream>>>(probs, win, wout, Wi, Wo);
  lora_stage1_kernel<<<dim3(NB_ * NH_), dim3(256), 0, stream>>>(x, Wi, Ht);
  lora_stage2_kernel<<<dim3(NB_ * NH_), dim3(256), 0, stream>>>(Ht, Wo, out);
}
